// ClassicMHCABlock_25512105738932
// MI455X (gfx1250) — matrix operand order repaired and checked by reading, not compiled, not run on hardware
//
#include <hip/hip_runtime.h>
#include <math.h>

// ---------------------------------------------------------------------------
// MI455X (gfx1250) classic MHCA block, bf16 WMMA everywhere.
// D=1024, H=16, DH=64, B=4, Nq=Nk=2048.
// Workspace layout (bf16 elements), total ~120 MB:
//   xb(16M) cb(16M) wq(2M) wk(2M) wv(2M) wo(2M) Q(16M) K(16M) V(16M) Vt(16M) O(16M)
// ---------------------------------------------------------------------------

typedef unsigned short u16;
typedef __attribute__((ext_vector_type(16))) __bf16 v16bf;
typedef __attribute__((ext_vector_type(8)))  float  v8f;
typedef __attribute__((ext_vector_type(4)))  unsigned int u32x4;

union AFrag { v16bf v; u32x4 u[2]; };
union Acc   { v8f   v; float f[8]; };

__device__ __forceinline__ u16 f2bf(float f) {
    unsigned int u = __float_as_uint(f);
    u += 0x7FFFu + ((u >> 16) & 1u);          // round-to-nearest-even
    return (u16)(u >> 16);
}
__device__ __forceinline__ float bf2f(u16 s) {
    return __uint_as_float(((unsigned int)s) << 16);
}

__device__ __forceinline__ v8f wmma_bf16(const AFrag& a, const AFrag& b, v8f c) {
    return __builtin_amdgcn_wmma_f32_16x16x32_bf16(false, a.v, false, b.v,
                                                   (short)0, c, false, false);
}

// ---------------------------------------------------------------------------
// fp32 -> bf16 conversion
// ---------------------------------------------------------------------------
__global__ void cvt_bf16_kernel(const float* __restrict__ src,
                                u16* __restrict__ dst, size_t n) {
    size_t i = (size_t)blockIdx.x * blockDim.x + threadIdx.x;
    if (i < n) dst[i] = f2bf(src[i]);
}

// ---------------------------------------------------------------------------
// C[m][n] = sum_k A[m][k] * W[n][k] + bias[n]     (M x 1024, K=N=1024)
// Block tile 128x128, 256 threads = 8 waves (4 along M x 2 along N),
// each wave: 32x64 via 2x4 WMMA tiles, K-step 32, LDS-staged operands.
// ---------------------------------------------------------------------------
template <bool OUT_F32>
__global__ __launch_bounds__(256)
void gemm_bf16_kernel(const u16* __restrict__ A, const u16* __restrict__ W,
                      const float* __restrict__ bias, void* __restrict__ Cout) {
    __shared__ __align__(16) u16 lA[128 * 32];   // 8 KB
    __shared__ __align__(16) u16 lB[128 * 32];   // 8 KB
    const int N = 1024;

    const int n0  = blockIdx.x * 128;
    const int m0  = blockIdx.y * 128;
    const int tid = threadIdx.x;
    const int lane = tid & 31;
    const int w   = tid >> 5;
    const int wm  = w & 3;        // row group of 32
    const int wn  = w >> 2;       // col group of 64
    const int l15 = lane & 15;
    const int hh  = lane >> 4;

    Acc acc[2][4];
#pragma unroll
    for (int i = 0; i < 2; i++)
#pragma unroll
        for (int j = 0; j < 4; j++)
#pragma unroll
            for (int r = 0; r < 8; r++) acc[i][j].f[r] = 0.0f;

    for (int k0 = 0; k0 < 1024; k0 += 32) {
        // Stage A tile and B tile (each 128x32 = 512 u32x4, 2 per thread).
#pragma unroll
        for (int rep = 0; rep < 2; rep++) {
            int idx = tid + rep * 256;
            int r = idx >> 2, q = idx & 3;
            ((u32x4*)lA)[idx] =
                ((const u32x4*)A)[(size_t)(m0 + r) * 128 + (k0 >> 3) + q];
            ((u32x4*)lB)[idx] =
                ((const u32x4*)W)[(size_t)(n0 + r) * 128 + (k0 >> 3) + q];
        }
        __syncthreads();

        AFrag af[2], bf[4];
#pragma unroll
        for (int mi = 0; mi < 2; mi++) {
            int row = 32 * wm + 16 * mi + l15;
            af[mi].u[0] = *(const u32x4*)&lA[row * 32 + hh * 8];
            af[mi].u[1] = *(const u32x4*)&lA[row * 32 + 16 + hh * 8];
        }
#pragma unroll
        for (int ni = 0; ni < 4; ni++) {
            int row = 64 * wn + 16 * ni + l15;
            bf[ni].u[0] = *(const u32x4*)&lB[row * 32 + hh * 8];
            bf[ni].u[1] = *(const u32x4*)&lB[row * 32 + 16 + hh * 8];
        }
#pragma unroll
        for (int mi = 0; mi < 2; mi++)
#pragma unroll
            for (int ni = 0; ni < 4; ni++)
                acc[mi][ni].v = wmma_bf16(af[mi], bf[ni], acc[mi][ni].v);
        __syncthreads();
    }

    // Epilogue: bias + store (C/D layout: lane n+16*(m/8), vgpr m%8).
#pragma unroll
    for (int mi = 0; mi < 2; mi++)
#pragma unroll
        for (int ni = 0; ni < 4; ni++) {
            int col = n0 + 64 * wn + 16 * ni + l15;
            float bv = bias[col];
#pragma unroll
            for (int r = 0; r < 8; r++) {
                int row = m0 + 32 * wm + 16 * mi + 8 * hh + r;
                float val = acc[mi][ni].f[r] + bv;
                if (OUT_F32)
                    ((float*)Cout)[(size_t)row * N + col] = val;
                else
                    ((u16*)Cout)[(size_t)row * N + col] = f2bf(val);
            }
        }
}

// ---------------------------------------------------------------------------
// In-place per-head L2 norm: one wave per (row, head) of 64 elements.
// ---------------------------------------------------------------------------
__global__ void qknorm_kernel(u16* __restrict__ Q, int total_rh) {
    int gw   = (int)((blockIdx.x * (size_t)blockDim.x + threadIdx.x) >> 5);
    int lane = threadIdx.x & 31;
    if (gw >= total_rh) return;
    int row = gw >> 4, h = gw & 15;
    u16* p = Q + (size_t)row * 1024 + h * 64 + lane * 2;
    float a = bf2f(p[0]), b = bf2f(p[1]);
    float ss = a * a + b * b;
#pragma unroll
    for (int m = 16; m >= 1; m >>= 1) ss += __shfl_xor(ss, m, 32);
    float inv = 1.0f / (sqrtf(ss) + 1e-6f);
    p[0] = f2bf(a * inv);
    p[1] = f2bf(b * inv);
}

// ---------------------------------------------------------------------------
// Vt[(b*16+h)*64 + d][n] = V[(b*2048+n)*1024 + h*64 + d]   (coalesced writes)
// ---------------------------------------------------------------------------
__global__ void vtrans_kernel(const u16* __restrict__ V, u16* __restrict__ Vt) {
    size_t i = (size_t)blockIdx.x * blockDim.x + threadIdx.x;
    if (i >= (size_t)4 * 2048 * 1024) return;
    int n = (int)(i & 2047);
    size_t t = i >> 11;
    int d = (int)(t & 63);  t >>= 6;
    int h = (int)(t & 15);  int b = (int)(t >> 4);
    Vt[i] = V[((size_t)(b * 2048 + n)) * 1024 + h * 64 + d];
}

// ---------------------------------------------------------------------------
// Flash attention: block = (b,h, 64 q rows), 4 waves, each wave owns a 16-row
// Q strip held in A-fragments. Per 64-row KV chunk: cooperative global->LDS
// staging of K and V^T (shared by all 4 waves, 4x traffic reduction), next-
// chunk global_prefetch, 8 WMMA for S, online softmax (16-lane butterflies),
// P relayout through a per-wave LDS slot, 8 WMMA for P*V into fp32 accums.
// Scale = 1.0 (QK-norm). LDS: 8KB K + 8KB V + 8KB P = 24KB.
// ---------------------------------------------------------------------------
__global__ __launch_bounds__(128)
void attn_kernel(const u16* __restrict__ Q, const u16* __restrict__ Kn,
                 const u16* __restrict__ Vt, u16* __restrict__ O) {
    __shared__ __align__(16) u16 lK[64 * 64];       // K chunk, [kv][dh]
    __shared__ __align__(16) u16 lV[64 * 64];       // V^T chunk, [dh][kv]
    __shared__ __align__(16) u16 lP[4 * 16 * 64];   // P slots, one per wave

    const int bh = blockIdx.x;            // 0..63
    const int b = bh >> 4, h = bh & 15;
    const int q0 = blockIdx.y * 64;
    const int tid = threadIdx.x;
    const int w = tid >> 5, lane = tid & 31;
    const int l15 = lane & 15, hh = lane >> 4;

    // Q A-fragments for this wave's 16-row strip (K dim = DH = 64 -> 2 frags)
    const int qrow = q0 + w * 16 + l15;
    const size_t qbase = ((size_t)(b * 2048 + qrow)) * 1024 + h * 64;
    AFrag qa[2];
    qa[0].u[0] = *(const u32x4*)(Q + qbase + hh * 8);
    qa[0].u[1] = *(const u32x4*)(Q + qbase + 16 + hh * 8);
    qa[1].u[0] = *(const u32x4*)(Q + qbase + 32 + hh * 8);
    qa[1].u[1] = *(const u32x4*)(Q + qbase + 48 + hh * 8);

    Acc o[4];
#pragma unroll
    for (int j = 0; j < 4; j++)
#pragma unroll
        for (int r = 0; r < 8; r++) o[j].f[r] = 0.0f;
    float rmax[8], rsum[8];
#pragma unroll
    for (int r = 0; r < 8; r++) { rmax[r] = -1e30f; rsum[r] = 0.0f; }

    const size_t kbase0 = ((size_t)(b * 2048)) * 1024 + h * 64;
    const size_t vtbase = ((size_t)((b * 16 + h) * 64)) * 2048;
    u16* myP = lP + w * 16 * 64;

    for (int c = 0; c < 2048; c += 64) {
        // ---- fetch this K / V^T chunk to registers (512 u32x4 each) ----
        u32x4 kreg[4], vreg[4];
#pragma unroll
        for (int rep = 0; rep < 4; rep++) {
            int idx = tid + rep * 128;
            int row = idx >> 3, q = idx & 7;
            kreg[rep] = *(const u32x4*)(Kn + kbase0 +
                                        (size_t)(c + row) * 1024 + q * 8);
            vreg[rep] = *(const u32x4*)(Vt + vtbase +
                                        (size_t)row * 2048 + c + q * 8);
        }
        // ---- prefetch next chunk while this one is consumed ----
        if (c + 64 < 2048) {
            __builtin_prefetch(Kn + kbase0 + (size_t)(c + 64 + (tid >> 1)) * 1024 +
                               (tid & 1) * 32, 0, 3);
            __builtin_prefetch(Vt + vtbase + (size_t)(tid >> 1) * 2048 +
                               (c + 64) + (tid & 1) * 32, 0, 3);
        }
        __syncthreads();   // all waves finished reading previous LDS chunk
#pragma unroll
        for (int rep = 0; rep < 4; rep++) {
            int idx = tid + rep * 128;
            ((u32x4*)lK)[idx] = kreg[rep];
            ((u32x4*)lV)[idx] = vreg[rep];
        }
        __syncthreads();   // staged chunk visible to all waves

        // ---- S = Q * K^T for this chunk (16 x 64 strip per wave) ----
        Acc s[4];
#pragma unroll
        for (int j = 0; j < 4; j++) {
            const u16* kr = &lK[(j * 16 + l15) * 64];
            AFrag b0, b1;
            b0.u[0] = *(const u32x4*)(kr + hh * 8);
            b0.u[1] = *(const u32x4*)(kr + 16 + hh * 8);
            b1.u[0] = *(const u32x4*)(kr + 32 + hh * 8);
            b1.u[1] = *(const u32x4*)(kr + 48 + hh * 8);
            v8f z = {};
            z = wmma_bf16(qa[0], b0, z);
            s[j].v = wmma_bf16(qa[1], b1, z);
        }

        // ---- online softmax ----
#pragma unroll
        for (int r = 0; r < 8; r++) {
            float mx = fmaxf(fmaxf(s[0].f[r], s[1].f[r]),
                             fmaxf(s[2].f[r], s[3].f[r]));
            mx = fmaxf(mx, __shfl_xor(mx, 1, 32));
            mx = fmaxf(mx, __shfl_xor(mx, 2, 32));
            mx = fmaxf(mx, __shfl_xor(mx, 4, 32));
            mx = fmaxf(mx, __shfl_xor(mx, 8, 32));
            float mnew = fmaxf(rmax[r], mx);
            float corr = __expf(rmax[r] - mnew);
            rmax[r] = mnew;
            float ps = 0.0f;
            const int m = r + 8 * hh;
#pragma unroll
            for (int j = 0; j < 4; j++) {
                float p = __expf(s[j].f[r] - mnew);
                ps += p;
                myP[m * 64 + j * 16 + l15] = f2bf(p);
            }
            ps += __shfl_xor(ps, 1, 32);
            ps += __shfl_xor(ps, 2, 32);
            ps += __shfl_xor(ps, 4, 32);
            ps += __shfl_xor(ps, 8, 32);
            rsum[r] = rsum[r] * corr + ps;
#pragma unroll
            for (int dj = 0; dj < 4; dj++) o[dj].f[r] *= corr;
        }
        // P writes/reads stay within this wave's private slot; per-wave DS
        // ordering (DScnt, in-order) makes them visible without a barrier.

        // ---- P as A-fragments (C-layout -> A-layout via LDS) ----
        AFrag pa[2];
        {
            const u16* pr = &myP[l15 * 64];
            pa[0].u[0] = *(const u32x4*)(pr + hh * 8);
            pa[0].u[1] = *(const u32x4*)(pr + 16 + hh * 8);
            pa[1].u[0] = *(const u32x4*)(pr + 32 + hh * 8);
            pa[1].u[1] = *(const u32x4*)(pr + 48 + hh * 8);
        }

        // ---- O += P * V (B-fragments from staged V^T rows) ----
#pragma unroll
        for (int dj = 0; dj < 4; dj++) {
            const u16* vr = &lV[(dj * 16 + l15) * 64];
            AFrag b0, b1;
            b0.u[0] = *(const u32x4*)(vr + hh * 8);
            b0.u[1] = *(const u32x4*)(vr + 16 + hh * 8);
            b1.u[0] = *(const u32x4*)(vr + 32 + hh * 8);
            b1.u[1] = *(const u32x4*)(vr + 48 + hh * 8);
            o[dj].v = wmma_bf16(pa[0], b0, o[dj].v);
            o[dj].v = wmma_bf16(pa[1], b1, o[dj].v);
        }
    }

    // ---- finalize: divide by row sums, store bf16 O in (B*Nq, D) layout ----
#pragma unroll
    for (int r = 0; r < 8; r++) {
        float inv = 1.0f / rsum[r];
        int row = q0 + w * 16 + 8 * hh + r;
        size_t ob = ((size_t)(b * 2048 + row)) * 1024 + h * 64;
#pragma unroll
        for (int dj = 0; dj < 4; dj++)
            O[ob + dj * 16 + l15] = f2bf(o[dj].f[r] * inv);
    }
}

// ---------------------------------------------------------------------------
extern "C" void kernel_launch(void* const* d_in, const int* in_sizes, int n_in,
                              void* d_out, int out_size, void* d_ws, size_t ws_size,
                              hipStream_t stream) {
    const float* x   = (const float*)d_in[0];
    const float* ctx = (const float*)d_in[1];
    const float* Wq  = (const float*)d_in[2];
    const float* bq  = (const float*)d_in[3];
    const float* Wk  = (const float*)d_in[4];
    const float* bk  = (const float*)d_in[5];
    const float* Wv  = (const float*)d_in[6];
    const float* bv  = (const float*)d_in[7];
    const float* Wo  = (const float*)d_in[8];
    const float* bo  = (const float*)d_in[9];
    float* out = (float*)d_out;

    const size_t MD = (size_t)8192 * 1024;   // activation elements
    const size_t WD = (size_t)1024 * 1024;   // weight elements
    u16* xb  = (u16*)d_ws;
    u16* cb  = xb  + MD;
    u16* wqb = cb  + MD;
    u16* wkb = wqb + WD;
    u16* wvb = wkb + WD;
    u16* wob = wvb + WD;
    u16* Qb  = wob + WD;
    u16* Kb  = Qb  + MD;
    u16* Vb  = Kb  + MD;
    u16* Vt  = Vb  + MD;
    u16* Ob  = Vt  + MD;

    // 1) fp32 -> bf16
    cvt_bf16_kernel<<<(unsigned)((MD + 255) / 256), 256, 0, stream>>>(x,   xb,  MD);
    cvt_bf16_kernel<<<(unsigned)((MD + 255) / 256), 256, 0, stream>>>(ctx, cb,  MD);
    cvt_bf16_kernel<<<(unsigned)((WD + 255) / 256), 256, 0, stream>>>(Wq,  wqb, WD);
    cvt_bf16_kernel<<<(unsigned)((WD + 255) / 256), 256, 0, stream>>>(Wk,  wkb, WD);
    cvt_bf16_kernel<<<(unsigned)((WD + 255) / 256), 256, 0, stream>>>(Wv,  wvb, WD);
    cvt_bf16_kernel<<<(unsigned)((WD + 255) / 256), 256, 0, stream>>>(Wo,  wob, WD);

    // 2) QKV projections (bf16 out); grid = (N/128, M/128)
    dim3 gg(1024 / 128, 8192 / 128);
    gemm_bf16_kernel<false><<<gg, 256, 0, stream>>>(xb, wqb, bq, Qb);
    gemm_bf16_kernel<false><<<gg, 256, 0, stream>>>(cb, wkb, bk, Kb);
    gemm_bf16_kernel<false><<<gg, 256, 0, stream>>>(cb, wvb, bv, Vb);

    // 3) QK norm (in place), 4) V transpose per (b,h)
    int total_rh = 8192 * 16;
    int nthr = total_rh * 32;
    qknorm_kernel<<<(nthr + 127) / 128, 128, 0, stream>>>(Qb, total_rh);
    qknorm_kernel<<<(nthr + 127) / 128, 128, 0, stream>>>(Kb, total_rh);
    vtrans_kernel<<<(unsigned)((MD + 255) / 256), 256, 0, stream>>>(Vb, Vt);

    // 5) attention: grid = (B*H, Nq/64)
    attn_kernel<<<dim3(64, 32), 128, 0, stream>>>(Qb, Kb, Vt, Ob);

    // 6) output projection (fp32 out)
    gemm_bf16_kernel<true><<<gg, 256, 0, stream>>>(Ob, wob, bo, out);
}
